// Decoder_MDCBlock1_88261577933063
// MI455X (gfx1250) — matrix operand order repaired and checked by reading, not compiled, not run on hardware
//
#include <hip/hip_runtime.h>

// ---------------------------------------------------------------------------
// CDNA5 (gfx1250) implicit-GEMM dynamic conv / deconv with bf16 WMMA.
// wave32; WMMA 16x16x32 bf16 -> f32 accum. Each wave computes MW 16x16 tiles
// (MW out-channel tiles x 16 pixels) sharing one B fragment per k-step.
// A tiles staged to LDS by the Tensor Data Mover (double-buffered).
// ---------------------------------------------------------------------------

typedef __bf16 bf16_t;
typedef __attribute__((ext_vector_type(16))) __bf16 v16bf;
typedef __attribute__((ext_vector_type(8)))  __bf16 v8bf;
typedef __attribute__((ext_vector_type(8)))  float  v8f;
typedef __attribute__((ext_vector_type(4))) unsigned int u32x4;
typedef __attribute__((ext_vector_type(8))) int i32x8;
typedef __attribute__((ext_vector_type(4))) int i32x4;

union U128 { uint4 u; v8bf v; };

#define KCHUNK 256   // A staging chunk in K (bf16 elems): 16*MW x 256 in LDS

#if __has_builtin(__builtin_amdgcn_tensor_load_to_lds) && \
    __has_builtin(__builtin_amdgcn_s_wait_tensorcnt)
#define USE_TDM 1
#else
#define USE_TDM 0
#endif

// ---------------------------------------------------------------------------
// Weight synthesis: W = w + tb*mb*att0 + tq*mq*att1 + tn*mn*att2 + tx*mx*att3
// conv layout   : Wout[((b*O+o)*I+i)*16 + (ky*4+kx)]
// deconv layout : Wout[((b*I+i)*O+o)*16 + ((3-ky)*4+(3-kx))]   (swap+flip)
// ---------------------------------------------------------------------------
__global__ __launch_bounds__(256)
void synth_weights(const float* __restrict__ w,
                   const float* __restrict__ tb, const float* __restrict__ tq,
                   const float* __restrict__ tn, const float* __restrict__ tx,
                   const float* __restrict__ mb, const float* __restrict__ mq,
                   const float* __restrict__ mn, const float* __restrict__ mx,
                   const float* __restrict__ att,
                   bf16_t* __restrict__ Wout, int O, int I, int B, int deconv)
{
    int idx = blockIdx.x * blockDim.x + threadIdx.x;
    int total = B * O * I * 16;
    if (idx >= total) return;
    int kk   = idx & 15;
    int rest = idx >> 4;
    int i = rest % I; rest /= I;
    int o = rest % O;
    int b = rest / O;

    int wi = (o * I + i) * 16 + kk;
    int mi = o * I + i;
    const float* ab = att + b * 4;
    float val = w[wi]
              + tb[wi] * mb[mi] * ab[0]
              + tq[wi] * mq[mi] * ab[1]
              + tn[wi] * mn[mi] * ab[2]
              + tx[wi] * mx[mi] * ab[3];

    size_t dst;
    if (!deconv) {
        dst = ((size_t)(b * O + o) * I + i) * 16 + kk;
    } else {
        int ky = kk >> 2, kx = kk & 3;
        int fk = (3 - ky) * 4 + (3 - kx);
        dst = ((size_t)(b * I + i) * O + o) * 16 + fk;
    }
    Wout[dst] = (bf16_t)val;
}

// ---------------------------------------------------------------------------
// Tensor Data Mover: 2-D tile load global->LDS (ISA ch.8 D# layout).
//   rows x rowlen bf16 elements, row stride in elements; LDS packed dense.
//   6-arg builtin variant: (g0:u32x4, g1:i32x8, g2:i32x4, g3:i32x4, pad:i32x8, cpol)
// ---------------------------------------------------------------------------
#if USE_TDM
__device__ inline void tdm_load_2d(unsigned int lds_off, const void* gptr,
                                   unsigned int rows, unsigned int rowlen,
                                   unsigned int row_stride)
{
    unsigned long long ga = (unsigned long long)(size_t)gptr;
    u32x4 g0;
    g0[0] = 1u;                                   // count=1, load, user mode
    g0[1] = lds_off;                              // lds_addr (bytes)
    g0[2] = (unsigned int)ga;                     // global_addr[31:0]
    g0[3] = (unsigned int)((ga >> 32) & 0x01FFFFFFu) | 0x80000000u; // [56:32]|type=2
    i32x8 g1;
    g1[0] = 0x00010000;                           // data_size=1 (2B), mask=0
    g1[1] = (int)(rowlen << 16);                  // tensor_dim0[15:0]
    g1[2] = (int)(rows << 16);                    // dim0 hi=0 | tensor_dim1[15:0]
    g1[3] = (int)(rowlen << 16);                  // dim1 hi=0 | tile_dim0
    g1[4] = (int)rows;                            // tile_dim1 | tile_dim2=0
    g1[5] = (int)row_stride;                      // tensor_dim0_stride[31:0]
    g1[6] = 0;
    g1[7] = 0;
    i32x4 gz4 = {0, 0, 0, 0};                     // groups 2/3 unused (2-D)
    i32x8 gz8 = {0, 0, 0, 0, 0, 0, 0, 0};
    __builtin_amdgcn_tensor_load_to_lds(g0, g1, gz4, gz4, gz8, 0);
}
#endif

// ---------------------------------------------------------------------------
// A-fragment from LDS (contiguous k order, same as gather_b; not the ISA table order):
//   lane m=lane&15, kbase=(lane<16)?0:16 ; elems[0..15]=K+kbase+j
// ---------------------------------------------------------------------------
__device__ inline v16bf load_a_frag(const bf16_t* As, int row, int k0, int kb_a)
{
    const bf16_t* ap = As + row * KCHUNK + k0 + kb_a;
    U128 lo, hi;
    lo.u = *(const uint4*)(ap);
    hi.u = *(const uint4*)(ap + 8);
    return __builtin_shufflevector(lo.v, hi.v,
           0,1,2,3,4,5,6,7, 8,9,10,11,12,13,14,15);
}

// ---------------------------------------------------------------------------
// B fragment gather: 16 taps of one input channel, offsets precomputed.
// ---------------------------------------------------------------------------
__device__ inline v16bf gather_b(const float* __restrict__ base, const int* off)
{
    v16bf bv;
    #pragma unroll
    for (int j = 0; j < 16; ++j) {
        float v = 0.f;
        if (off[j] >= 0) v = base[off[j]];
        bv[j] = (bf16_t)v;
    }
    return bv;
}

// ---------------------------------------------------------------------------
// Shared GEMM mainloop: stages A (16*MW x KCHUNK) via TDM double-buffer,
// accumulates MW 16x16 tiles against the gathered B fragment.
// ---------------------------------------------------------------------------
template <int MW>
__device__ inline void gemm_mainloop(v8f* acc, bf16_t (*As)[16 * MW * KCHUNK],
                                     const bf16_t* __restrict__ Wb,
                                     const float* __restrict__ xb,
                                     const int* off, int Kred, int hs,
                                     int wave, int kb_a, int kb_b, int m)
{
#if USE_TDM
    if (wave == 0)
        tdm_load_2d((unsigned int)(size_t)&As[0][0], Wb, 16 * MW, KCHUNK,
                    (unsigned int)Kred);
    int cur = 0;
    for (int kc = 0; kc < Kred; kc += KCHUNK, cur ^= 1) {
        const bool has_next = (kc + KCHUNK) < Kred;
        if (wave == 0) {
            if (has_next) {
                tdm_load_2d((unsigned int)(size_t)&As[cur ^ 1][0],
                            Wb + (size_t)(kc + KCHUNK), 16 * MW, KCHUNK,
                            (unsigned int)Kred);
                __builtin_amdgcn_s_wait_tensorcnt(1);  // oldest (current buf) done
            } else {
                __builtin_amdgcn_s_wait_tensorcnt(0);
            }
        }
        __syncthreads();
        const bf16_t* Ab = &As[cur][0];
        for (int k0 = 0; k0 < KCHUNK; k0 += 32) {
            const int ci = (kc + k0 + kb_b) >> 4;
            const float* base = xb + ((size_t)ci << (2 * hs));
            v16bf bv = gather_b(base, off);
            #pragma unroll
            for (int t = 0; t < MW; ++t) {
                v16bf a = load_a_frag(Ab, t * 16 + m, k0, kb_a);
                acc[t] = __builtin_amdgcn_wmma_f32_16x16x32_bf16(
                             false, a, false, bv, (short)0, acc[t], false, false);
            }
        }
        __syncthreads();   // all readers done before this buffer is re-filled
    }
#else
    (void)wave;
    for (int kc = 0; kc < Kred; kc += KCHUNK) {
        __syncthreads();
        {
            int t   = threadIdx.x;
            int col = (t & 15) << 4;
            #pragma unroll
            for (int s = 0; s < MW; ++s) {
                int row = (t >> 4) + 16 * s;
                const uint4* s4 = (const uint4*)(Wb + (size_t)row * Kred + kc + col);
                uint4 v0 = s4[0], v1 = s4[1];
                uint4* d4 = (uint4*)&As[0][row * KCHUNK + col];
                d4[0] = v0; d4[1] = v1;
            }
        }
        __syncthreads();
        for (int k0 = 0; k0 < KCHUNK; k0 += 32) {
            const int ci = (kc + k0 + kb_b) >> 4;
            const float* base = xb + ((size_t)ci << (2 * hs));
            v16bf bv = gather_b(base, off);
            #pragma unroll
            for (int t = 0; t < MW; ++t) {
                v16bf a = load_a_frag(&As[0][0], t * 16 + m, k0, kb_a);
                acc[t] = __builtin_amdgcn_wmma_f32_16x16x32_bf16(
                             false, a, false, bv, (short)0, acc[t], false, false);
            }
        }
    }
#endif
}

// ---------------------------------------------------------------------------
// Dynamic conv: y = PReLU(conv_s2_p1_k4(x, W) + bias) [- sub]
// hs = log2(Hin). MW out-channel tiles per wave. Kred = Cin*16.
// ---------------------------------------------------------------------------
template <int MW>
__global__ __launch_bounds__(256)
void dyn_conv_wmma(const float* __restrict__ x, const bf16_t* __restrict__ Wg,
                   const float* __restrict__ bias, const float* __restrict__ aP,
                   const float* __restrict__ sub, float* __restrict__ out,
                   int Cin, int hs, int Cout)
{
    const int b    = blockIdx.z;
    const int mt   = blockIdx.y;                 // block of 16*MW out channels
    const int wave = threadIdx.x >> 5;
    const int lane = threadIdx.x & 31;
    const int Hin  = 1 << hs;
    const int Ho   = Hin >> 1;
    const int Npix = 1 << (2 * hs - 2);
    const int nt   = blockIdx.x * 8 + wave;
    const int Kred = Cin << 4;

    const bf16_t* Wb = Wg + ((size_t)b * Cout + mt * 16 * MW) * (size_t)Kred;
    const float*  xb = x + (((size_t)b * Cin) << (2 * hs));

    const int n  = lane & 15;
    const int p  = nt * 16 + n;
    const int py = p >> (hs - 1);
    const int px = p & (Ho - 1);
    const int kb_b = (lane >> 4) << 4;   // 0 or 16
    const int kb_a = (lane >> 4) << 4;   // 0 or 16
    const int m    = lane & 15;

    // k-invariant tap offsets (within one channel plane), -1 = zero-pad
    int off[16];
    #pragma unroll
    for (int ky = 0; ky < 4; ++ky) {
        int iy = 2 * py + ky - 1;
        bool rok = (unsigned)iy < (unsigned)Hin;
        #pragma unroll
        for (int kx = 0; kx < 4; ++kx) {
            int ix = 2 * px + kx - 1;
            bool cok = (unsigned)ix < (unsigned)Hin;
            off[ky * 4 + kx] = (rok && cok) ? ((iy << hs) + ix) : -1;
        }
    }

    v8f acc[MW];
    #pragma unroll
    for (int t = 0; t < MW; ++t) acc[t] = v8f{0.f,0.f,0.f,0.f,0.f,0.f,0.f,0.f};

#if USE_TDM
    __shared__ __align__(16) bf16_t As[2][16 * MW * KCHUNK];
#else
    __shared__ __align__(16) bf16_t As[1][16 * MW * KCHUNK];
#endif
    gemm_mainloop<MW>(acc, As, Wb, xb, off, Kred, hs, wave, kb_a, kb_b, m);

    const float alpha = aP[0];
    float* ob = out + (size_t)b * Cout * Npix;
    const float* sb = sub ? sub + (size_t)b * Cout * Npix : nullptr;
    #pragma unroll
    for (int t = 0; t < MW; ++t) {
        #pragma unroll
        for (int r = 0; r < 8; ++r) {
            int mm = r + ((lane >> 4) << 3);
            int o  = mt * 16 * MW + t * 16 + mm;
            float y = acc[t][r] + bias[o];
            y = (y >= 0.f) ? y : alpha * y;
            if (sb) y -= sb[(size_t)o * Npix + p];
            ob[(size_t)o * Npix + p] = y;
        }
    }
}

// ---------------------------------------------------------------------------
// Dynamic deconv (stride-2 upsample, k4, pad=2), weights pre-swapped+flipped.
// y[o,Y,X] = sum Wd[o,ci,ky,kx] * x[ci,(Y+ky-2)/2,(X+kx-2)/2]  (parity taps)
// hs = log2(Hin input spatial). Epilogue: PReLU(+bias) [+ residual]
// ---------------------------------------------------------------------------
template <int MW>
__global__ __launch_bounds__(256)
void dyn_deconv_wmma(const float* __restrict__ x, const bf16_t* __restrict__ Wg,
                     const float* __restrict__ bias, const float* __restrict__ aP,
                     const float* __restrict__ res, float* __restrict__ out,
                     int Cin, int hs, int Cout)
{
    const int b    = blockIdx.z;
    const int mt   = blockIdx.y;
    const int wave = threadIdx.x >> 5;
    const int lane = threadIdx.x & 31;
    const int Hin  = 1 << hs;
    const int Hout = Hin << 1;
    const int Npix = 1 << (2 * hs + 2);
    const int nt   = blockIdx.x * 8 + wave;
    const int Kred = Cin << 4;

    const bf16_t* Wb = Wg + ((size_t)b * Cout + mt * 16 * MW) * (size_t)Kred;
    const float*  xb = x + (((size_t)b * Cin) << (2 * hs));

    const int n  = lane & 15;
    const int p  = nt * 16 + n;
    const int py = p >> (hs + 1);
    const int px = p & (Hout - 1);
    const int kb_b = (lane >> 4) << 4;
    const int kb_a = (lane >> 4) << 4;
    const int m    = lane & 15;

    int off[16];
    #pragma unroll
    for (int ky = 0; ky < 4; ++ky) {
        int sy = py + ky - 2;
        bool rok = (sy >= 0) && !(sy & 1) && ((sy >> 1) < Hin);
        int iy = sy >> 1;
        #pragma unroll
        for (int kx = 0; kx < 4; ++kx) {
            int sx = px + kx - 2;
            bool cok = (sx >= 0) && !(sx & 1) && ((sx >> 1) < Hin);
            int ix = sx >> 1;
            off[ky * 4 + kx] = (rok && cok) ? ((iy << hs) + ix) : -1;
        }
    }

    v8f acc[MW];
    #pragma unroll
    for (int t = 0; t < MW; ++t) acc[t] = v8f{0.f,0.f,0.f,0.f,0.f,0.f,0.f,0.f};

#if USE_TDM
    __shared__ __align__(16) bf16_t As[2][16 * MW * KCHUNK];
#else
    __shared__ __align__(16) bf16_t As[1][16 * MW * KCHUNK];
#endif
    gemm_mainloop<MW>(acc, As, Wb, xb, off, Kred, hs, wave, kb_a, kb_b, m);

    const float alpha = aP[0];
    float* ob = out + (size_t)b * Cout * Npix;
    const float* rb = res ? res + (size_t)b * Cout * Npix : nullptr;
    #pragma unroll
    for (int t = 0; t < MW; ++t) {
        #pragma unroll
        for (int r = 0; r < 8; ++r) {
            int mm = r + ((lane >> 4) << 3);
            int o  = mt * 16 * MW + t * 16 + mm;
            float y = acc[t][r] + bias[o];
            y = (y >= 0.f) ? y : alpha * y;
            if (rb) y += rb[(size_t)o * Npix + p];
            ob[(size_t)o * Npix + p] = y;
        }
    }
}

// ---------------------------------------------------------------------------
// Host orchestration
// ---------------------------------------------------------------------------
static inline int ilog2(int v) { int s = 0; while ((1 << s) < v) ++s; return s; }

extern "C" void kernel_launch(void* const* d_in, const int* in_sizes, int n_in,
                              void* d_out, int out_size, void* d_ws, size_t ws_size,
                              hipStream_t stream)
{
    (void)in_sizes; (void)n_in; (void)out_size; (void)ws_size;
    const int B = 8, NF = 32;

    const float* ft_h  = (const float*)d_in[0];
    const float* ft_l0 = (const float*)d_in[1];
    const float* ft_l1 = (const float*)d_in[2];
    const float* ft_l2 = (const float*)d_in[3];
    const float* att   = (const float*)d_in[4];

    // param leaves: blocks 0..2 = down[0..2], 3..5 = up[0..2]
    // leaf order (sorted keys): 0=a 1=b 2=mb 3=mn 4=mq 5=mx 6=tb 7=tn 8=tq 9=tx 10=w
    const float* P[6][11];
    for (int blk = 0; blk < 6; ++blk)
        for (int k = 0; k < 11; ++k)
            P[blk][k] = (const float*)d_in[5 + blk * 11 + k];

    // ---- workspace carve ----
    char* wp = (char*)d_ws;
    auto carve = [&](size_t bytes) -> char* {
        char* r = wp;
        wp += (bytes + 255) & ~(size_t)255;
        return r;
    };
    bf16_t* Wc[3]; bf16_t* Wd[3];
    for (int j = 0; j < 3; ++j) {
        size_t e = (size_t)B * (NF << (j + 1)) * (NF << j) * 16;
        Wc[j] = (bf16_t*)carve(e * sizeof(bf16_t));
    }
    for (int j = 0; j < 3; ++j) {
        size_t e = (size_t)B * (NF << (j + 1)) * (NF << j) * 16;
        Wd[j] = (bf16_t*)carve(e * sizeof(bf16_t));
    }
    float* tmp0 = (float*)carve((size_t)B * 64 * 64 * 64 * sizeof(float));   // 8 MB
    float* tmp1 = (float*)carve((size_t)B * 64 * 64 * 64 * sizeof(float));   // 8 MB
    float* fus0 = (float*)carve((size_t)B * 32 * 128 * 128 * sizeof(float)); // 16 MB
    float* fus1 = (float*)carve((size_t)B * 32 * 128 * 128 * sizeof(float)); // 16 MB

    // ---- weight synthesis ----
    auto synth = [&](int blk, bf16_t* Wout, int O, int I, int dec) {
        int total = B * O * I * 16;
        synth_weights<<<(total + 255) / 256, 256, 0, stream>>>(
            P[blk][10], P[blk][6], P[blk][8], P[blk][7], P[blk][9],
            P[blk][2],  P[blk][4], P[blk][3], P[blk][5], att,
            Wout, O, I, B, dec);
    };
    for (int j = 0; j < 3; ++j) synth(j,     Wc[j], NF << (j + 1), NF << j, 0);
    for (int j = 0; j < 3; ++j) synth(3 + j, Wd[j], NF << (j + 1), NF << j, 1);

    // ---- block launchers ----
    auto conv = [&](const float* x, const bf16_t* W, int pb,
                    const float* sub, float* o, int Cin, int Hin, int Cout) {
        int hs = ilog2(Hin);
        int Ho = Hin >> 1;
        dim3 g((Ho * Ho) / 128, Cout / 64, B);          // MW = 4
        dyn_conv_wmma<4><<<g, 256, 0, stream>>>(x, W, P[pb][1], P[pb][0], sub, o,
                                                Cin, hs, Cout);
    };
    auto deconv = [&](const float* x, const bf16_t* W, int pb,
                      const float* res, float* o, int Cin, int Hin, int Cout) {
        int hs = ilog2(Hin);
        int Hout = Hin << 1;
        dim3 g((Hout * Hout) / 128, Cout / 32, B);      // MW = 2
        dyn_deconv_wmma<2><<<g, 256, 0, stream>>>(x, W, P[pb][1], P[pb][0], res, o,
                                                  Cin, hs, Cout);
    };

    // ---- i = 0 : down0,down1,down2 ; -ft_l0 ; up2,up1,up0 ; +ft_h -> fus0 ----
    conv  (ft_h, Wc[0], 0, nullptr, tmp0,  32, 128,  64);
    conv  (tmp0, Wc[1], 1, nullptr, tmp1,  64,  64, 128);
    conv  (tmp1, Wc[2], 2, ft_l0,   tmp0, 128,  32, 256);
    deconv(tmp0, Wd[2], 5, nullptr, tmp1, 256,  16, 128);
    deconv(tmp1, Wd[1], 4, nullptr, tmp0, 128,  32,  64);
    deconv(tmp0, Wd[0], 3, ft_h,    fus0,  64,  64,  32);

    // ---- i = 1 : down0,down1 ; -ft_l1 ; up1,up0 ; +fus0 -> fus1 ----
    conv  (fus0, Wc[0], 0, nullptr, tmp0,  32, 128,  64);
    conv  (tmp0, Wc[1], 1, ft_l1,   tmp1,  64,  64, 128);
    deconv(tmp1, Wd[1], 4, nullptr, tmp0, 128,  32,  64);
    deconv(tmp0, Wd[0], 3, fus0,    fus1,  64,  64,  32);

    // ---- i = 2 : down0 ; -ft_l2 ; up0 ; +fus1 -> d_out ----
    conv  (fus1, Wc[0], 0, ft_l2,   tmp0,  32, 128,  64);
    deconv(tmp0, Wd[0], 3, fus1, (float*)d_out, 64, 64, 32);
}
